// AxialAttention_69295002354053
// MI455X (gfx1250) — matrix operand order repaired and checked by reading, not compiled, not run on hardware
//
#include <hip/hip_runtime.h>
#include <hip/hip_bf16.h>
#include <math.h>

#define H 8
#define DH 32
#define DN 256
#define DE 128
#define RDIM 128
#define WW 256
#define TOK (RDIM * WW)      // 32768 tokens
#define NEGVAL (-1.0e30f)

typedef __attribute__((ext_vector_type(16))) __bf16        v16bf;
typedef __attribute__((ext_vector_type(8)))  float         v8f;
typedef __attribute__((ext_vector_type(16))) unsigned short v16u;

union frag_u { v16u u; v16bf b; uint4 q[2]; };

__device__ __forceinline__ unsigned short f2bf(float f) {
  unsigned int u = __float_as_uint(f);
  unsigned int r = (u + 0x7FFFu + ((u >> 16) & 1u)) >> 16;
  return (unsigned short)r;
}

// ---- WMMA fragment loaders (CDNA5 16x16x32 bf16 layouts, wave32) ----
// A: 16(M) x 32(K), row-major source, leading dim ld (elements).
// Lane L: m = L&15, half = L>>4. Elements 0..7 -> K = 8*half+0..7; 8..15 -> K = 16+8*half+0..7.
__device__ __forceinline__ frag_u load_a_frag(const unsigned short* base, int ld, int lane) {
  int m = lane & 15, hf = lane >> 4;
  const unsigned short* rp = base + (size_t)m * ld + hf * 8;
  frag_u f;
  f.q[0] = *(const uint4*)(rp);
  f.q[1] = *(const uint4*)(rp + 16);
  return f;
}
// B: 32(K) x 16(N); source has each column's K contiguous, column stride ldc.
// Lane L: n = L&15, half = L>>4. Elements 0..7 -> K = 8*half+0..7; 8..15 -> K = 16+8*half+0..7.
__device__ __forceinline__ frag_u load_b_frag(const unsigned short* colbase, int ldc, int lane) {
  int n = lane & 15, hf = lane >> 4;
  const unsigned short* cp = colbase + (size_t)n * ldc + hf * 8;
  frag_u f;
  f.q[0] = *(const uint4*)(cp);
  f.q[1] = *(const uint4*)(cp + 16);
  return f;
}

__device__ __forceinline__ float red_sum32(float v) {
  #pragma unroll
  for (int m = 16; m >= 1; m >>= 1) v += __shfl_xor(v, m, 32);
  return v;
}
__device__ __forceinline__ float red_sum16(float v) {
  #pragma unroll
  for (int m = 8; m >= 1; m >>= 1) v += __shfl_xor(v, m, 16);
  return v;
}
__device__ __forceinline__ float red_max16(float v) {
  #pragma unroll
  for (int m = 8; m >= 1; m >>= 1) v = fmaxf(v, __shfl_xor(v, m, 16));
  return v;
}

// ---------------------------------------------------------------------------
// Kernel 1: weights -> bf16, transposed to [out_col][k] for B-fragment loads.
// WT rows: 0..255 = Wq cols, 256..767 = Wkv cols, 768..1023 = Wg cols.
// ---------------------------------------------------------------------------
__global__ __launch_bounds__(256)
void wconv_kernel(const float* __restrict__ Wq, const float* __restrict__ Wkv,
                  const float* __restrict__ Wg, const float* __restrict__ Wo,
                  unsigned short* __restrict__ WT, unsigned short* __restrict__ WoT) {
  int n = blockIdx.x;     // 0..1279
  int k = threadIdx.x;    // 0..255
  if (n < 1024) {
    float v;
    if (n < 256)      v = Wq [k * 256 + n];
    else if (n < 768) v = Wkv[k * 512 + (n - 256)];
    else              v = Wg [k * 256 + (n - 768)];
    WT[(size_t)n * 256 + k] = f2bf(v);
  } else {
    int nn = n - 1024;
    WoT[(size_t)nn * 256 + k] = f2bf(Wo[k * 256 + nn]);
  }
}

// ---------------------------------------------------------------------------
// Kernel 2: node LayerNorm -> bf16 xn[TOK][256]. One wave per token.
// ---------------------------------------------------------------------------
__global__ __launch_bounds__(256)
void ln_nodes_kernel(const float* __restrict__ x, const float* __restrict__ g,
                     const float* __restrict__ b, unsigned short* __restrict__ xn) {
  int wid = threadIdx.x >> 5, lane = threadIdx.x & 31;
  int t = blockIdx.x * 8 + wid;
  const float* xr = x + (size_t)t * DN;
  float4 a0 = ((const float4*)xr)[lane * 2 + 0];
  float4 a1 = ((const float4*)xr)[lane * 2 + 1];
  float vals[8] = {a0.x, a0.y, a0.z, a0.w, a1.x, a1.y, a1.z, a1.w};
  float s = 0.f, s2 = 0.f;
  #pragma unroll
  for (int i = 0; i < 8; i++) { s += vals[i]; s2 += vals[i] * vals[i]; }
  s = red_sum32(s); s2 = red_sum32(s2);
  float mu = s * (1.f / 256.f);
  float var = s2 * (1.f / 256.f) - mu * mu;
  float inv = rsqrtf(var + 1e-5f);
  int c0 = lane * 8;
  unsigned int pk[4];
  #pragma unroll
  for (int jj = 0; jj < 4; jj++) {
    unsigned short lo = f2bf((vals[2 * jj + 0] - mu) * inv * g[c0 + 2 * jj + 0] + b[c0 + 2 * jj + 0]);
    unsigned short hi = f2bf((vals[2 * jj + 1] - mu) * inv * g[c0 + 2 * jj + 1] + b[c0 + 2 * jj + 1]);
    pk[jj] = (unsigned int)lo | ((unsigned int)hi << 16);
  }
  *(uint4*)(xn + (size_t)t * DN + c0) = make_uint4(pk[0], pk[1], pk[2], pk[3]);
}

// ---------------------------------------------------------------------------
// Kernel 3: edge LayerNorm + bias[h][i][j] = LN(edges[i,j,:]) . W_edge[:,h],
// with edge-mask fill. One wave per (i,j) pair.
// ---------------------------------------------------------------------------
__global__ __launch_bounds__(256)
void edge_bias_kernel(const float* __restrict__ edges, const float* __restrict__ lg,
                      const float* __restrict__ lb, const float* __restrict__ We,
                      const unsigned char* __restrict__ edge_mask, float* __restrict__ bias) {
  int wid = threadIdx.x >> 5, lane = threadIdx.x & 31;
  int p = blockIdx.x * 8 + wid;                 // 0..65535 = i*256+j
  const float* er = edges + (size_t)p * DE;
  float4 a = ((const float4*)er)[lane];
  float vals[4] = {a.x, a.y, a.z, a.w};
  float s = vals[0] + vals[1] + vals[2] + vals[3];
  float s2 = vals[0]*vals[0] + vals[1]*vals[1] + vals[2]*vals[2] + vals[3]*vals[3];
  s = red_sum32(s); s2 = red_sum32(s2);
  float mu = s * (1.f / 128.f);
  float var = s2 * (1.f / 128.f) - mu * mu;
  float inv = rsqrtf(var + 1e-5f);
  int c0 = lane * 4;
  float xr[4];
  #pragma unroll
  for (int c = 0; c < 4; c++) xr[c] = (vals[c] - mu) * inv * lg[c0 + c] + lb[c0 + c];
  float hacc[H];
  #pragma unroll
  for (int hh = 0; hh < H; hh++) {
    float t = 0.f;
    #pragma unroll
    for (int c = 0; c < 4; c++) t += xr[c] * We[(size_t)(c0 + c) * H + hh];
    hacc[hh] = red_sum32(t);
  }
  if (lane < H) {
    float v = hacc[0];
    #pragma unroll
    for (int hh = 1; hh < H; hh++) v = (lane == hh) ? hacc[hh] : v;
    bias[(size_t)lane * (WW * WW) + p] = edge_mask[p] ? v : NEGVAL;
  }
}

// ---------------------------------------------------------------------------
// Kernel 4: fused QKV+gate GEMM: [32768 x 256](bf16) x [256 x 1024](bf16).
// One wave per 16x64 output block: A-frag loaded once per K-step, reused for
// 4 B-frags / 4 WMMAs (1.25 frag loads per WMMA instead of 2).
// Epilogue scatters: Q/K -> [r][h][t][dh] bf16, V -> [r][h][dh][t] bf16,
// gates -> sigmoid(.+bg) f32 [t][256].
// ---------------------------------------------------------------------------
__global__ __launch_bounds__(256)
void qkvg_kernel(const unsigned short* __restrict__ xn, const unsigned short* __restrict__ WT,
                 const float* __restrict__ bg,
                 unsigned short* __restrict__ Qb, unsigned short* __restrict__ Kb,
                 unsigned short* __restrict__ Vt, float* __restrict__ gates) {
  int wid = threadIdx.x >> 5, lane = threadIdx.x & 31;
  int blk = blockIdx.x * 8 + wid;              // 2048 m-tiles x 16 n-blocks(64)
  int nb = blk & 15, mtile = blk >> 4;
  int m0 = mtile * 16, nbase = nb * 64;
  v8f acc[4];
  #pragma unroll
  for (int tt = 0; tt < 4; tt++) acc[tt] = (v8f){0.f,0.f,0.f,0.f,0.f,0.f,0.f,0.f};
  #pragma unroll
  for (int kk = 0; kk < 256; kk += 32) {
    frag_u a = load_a_frag(xn + (size_t)m0 * 256 + kk, 256, lane);
    #pragma unroll
    for (int tt = 0; tt < 4; tt++) {
      frag_u bf = load_b_frag(WT + (size_t)(nbase + tt * 16) * 256 + kk, 256, lane);
      acc[tt] = __builtin_amdgcn_wmma_f32_16x16x32_bf16(false, a.b, false, bf.b,
                                                        (short)0, acc[tt], false, false);
    }
  }
  int n = lane & 15, hf = lane >> 4;
  #pragma unroll
  for (int tt = 0; tt < 4; tt++) {
    int n0 = nbase + tt * 16;
    int col = n0 + n;
    if (n0 < 256) {                                       // Q
      int h = col >> 5, d = col & 31;
      #pragma unroll
      for (int i = 0; i < 8; i++) {
        int t = m0 + i + 8 * hf; int r = t >> 8, tw = t & 255;
        Qb[(((size_t)(r * H + h)) * WW + tw) * DH + d] = f2bf(acc[tt][i]);
      }
    } else if (n0 < 512) {                                // K
      int c = col - 256; int h = c >> 5, d = c & 31;
      #pragma unroll
      for (int i = 0; i < 8; i++) {
        int t = m0 + i + 8 * hf; int r = t >> 8, tw = t & 255;
        Kb[(((size_t)(r * H + h)) * WW + tw) * DH + d] = f2bf(acc[tt][i]);
      }
    } else if (n0 < 768) {                                // V (transposed)
      int c = col - 512; int h = c >> 5, d = c & 31;
      #pragma unroll
      for (int i = 0; i < 8; i++) {
        int t = m0 + i + 8 * hf; int r = t >> 8, tw = t & 255;
        Vt[(((size_t)(r * H + h)) * DH + d) * WW + tw] = f2bf(acc[tt][i]);
      }
    } else {                                              // gates
      int c = col - 768;
      float bgv = bg[c];
      #pragma unroll
      for (int i = 0; i < 8; i++) {
        int t = m0 + i + 8 * hf;
        gates[(size_t)t * 256 + c] = 1.f / (1.f + expf(-(acc[tt][i] + bgv)));
      }
    }
  }
}

// ---------------------------------------------------------------------------
// Kernel 5: flash row-attention. Block = (r,h), 16 waves, wave = 16 queries.
// S tile via WMMA, bias + mask, online softmax over 32-key chunks,
// P staged through LDS (C-layout -> A-layout), PV via WMMA, gating epilogue.
// ---------------------------------------------------------------------------
__global__ __launch_bounds__(512)
void attn_kernel(const unsigned short* __restrict__ Qb, const unsigned short* __restrict__ Kb,
                 const unsigned short* __restrict__ Vt, const float* __restrict__ bias,
                 const unsigned char* __restrict__ mask, const float* __restrict__ gates,
                 unsigned short* __restrict__ G) {
  __shared__ unsigned short lp[16][16][32];    // per-wave 16x32 P tile (bf16)
  int wid = threadIdx.x >> 5, lane = threadIdx.x & 31;
  int r = blockIdx.x >> 3, h = blockIdx.x & 7;
  int q0 = wid * 16;
  int n = lane & 15, hf = lane >> 4;
  size_t rh = (size_t)(r * H + h);

  frag_u aq = load_a_frag(Qb + (rh * WW + q0) * DH, DH, lane);
  const float scale = 0.17677669529663687f;    // 1/sqrt(32)

  float mrun[8], lrun[8], qm[8];
  int rowtok[8];
  v8f o0 = {0.f,0.f,0.f,0.f,0.f,0.f,0.f,0.f}, o1 = o0;
  #pragma unroll
  for (int i = 0; i < 8; i++) {
    rowtok[i] = q0 + i + 8 * hf;
    qm[i] = mask[r * WW + rowtok[i]] ? 1.f : 0.f;
    mrun[i] = -3.0e38f; lrun[i] = 0.f;
  }
  const float* biasrow = bias + (size_t)h * WW * WW;
  const v8f zero = {0.f,0.f,0.f,0.f,0.f,0.f,0.f,0.f};

  #pragma unroll 1
  for (int jt = 0; jt < WW; jt += 32) {
    frag_u bk0 = load_b_frag(Kb + (rh * WW + jt) * DH, DH, lane);
    frag_u bk1 = load_b_frag(Kb + (rh * WW + jt + 16) * DH, DH, lane);
    v8f s0 = __builtin_amdgcn_wmma_f32_16x16x32_bf16(false, aq.b, false, bk0.b,
                                                     (short)0, zero, false, false);
    v8f s1 = __builtin_amdgcn_wmma_f32_16x16x32_bf16(false, aq.b, false, bk1.b,
                                                     (short)0, zero, false, false);
    float km0 = mask[r * WW + jt + n] ? 1.f : 0.f;
    float km1 = mask[r * WW + jt + 16 + n] ? 1.f : 0.f;
    float f0[8], f1[8];
    #pragma unroll
    for (int i = 0; i < 8; i++) {
      float b0 = biasrow[(size_t)rowtok[i] * WW + jt + n];
      float b1 = biasrow[(size_t)rowtok[i] * WW + jt + 16 + n];
      f0[i] = (qm[i] * km0 > 0.f) ? (s0[i] * scale + b0) : NEGVAL;
      f1[i] = (qm[i] * km1 > 0.f) ? (s1[i] * scale + b1) : NEGVAL;
    }
    #pragma unroll
    for (int i = 0; i < 8; i++) {
      float rmax = red_max16(fmaxf(f0[i], f1[i]));
      float mnew = fmaxf(mrun[i], rmax);
      float alpha = expf(mrun[i] - mnew);
      float p0 = expf(f0[i] - mnew);
      float p1 = expf(f1[i] - mnew);
      float rs = red_sum16(p0 + p1);
      lrun[i] = lrun[i] * alpha + rs;
      mrun[i] = mnew;
      o0[i] *= alpha; o1[i] *= alpha;
      lp[wid][i + 8 * hf][n]      = f2bf(p0);
      lp[wid][i + 8 * hf][16 + n] = f2bf(p1);
    }
    asm volatile("s_wait_dscnt 0" ::: "memory");  // same-wave DS in-order; make P visible
    frag_u ap  = load_a_frag(&lp[wid][0][0], 32, lane);
    frag_u bv0 = load_b_frag(Vt + (rh * DH + 0)  * WW + jt, WW, lane);
    frag_u bv1 = load_b_frag(Vt + (rh * DH + 16) * WW + jt, WW, lane);
    o0 = __builtin_amdgcn_wmma_f32_16x16x32_bf16(false, ap.b, false, bv0.b,
                                                 (short)0, o0, false, false);
    o1 = __builtin_amdgcn_wmma_f32_16x16x32_bf16(false, ap.b, false, bv1.b,
                                                 (short)0, o1, false, false);
  }
  #pragma unroll
  for (int i = 0; i < 8; i++) {
    float invl = 1.f / lrun[i];
    size_t t = (size_t)(r * WW + rowtok[i]);
    float g0 = gates[t * 256 + h * 32 + n];
    float g1 = gates[t * 256 + h * 32 + 16 + n];
    G[t * 256 + h * 32 + n]      = f2bf(o0[i] * invl * g0);
    G[t * 256 + h * 32 + 16 + n] = f2bf(o1[i] * invl * g1);
  }
}

// ---------------------------------------------------------------------------
// Kernel 6: output projection [32768 x 256](bf16) x WoT + bo -> f32 d_out.
// One wave per 16x64 output block (A-frag reused across 4 WMMAs per K-step).
// ---------------------------------------------------------------------------
__global__ __launch_bounds__(256)
void out_gemm_kernel(const unsigned short* __restrict__ G, const unsigned short* __restrict__ WoT,
                     const float* __restrict__ bo, float* __restrict__ out) {
  int wid = threadIdx.x >> 5, lane = threadIdx.x & 31;
  int blk = blockIdx.x * 8 + wid;              // 2048 m-tiles x 4 n-blocks(64)
  int nb = blk & 3, mtile = blk >> 2;
  int m0 = mtile * 16, nbase = nb * 64;
  v8f acc[4];
  #pragma unroll
  for (int tt = 0; tt < 4; tt++) acc[tt] = (v8f){0.f,0.f,0.f,0.f,0.f,0.f,0.f,0.f};
  #pragma unroll
  for (int kk = 0; kk < 256; kk += 32) {
    frag_u a = load_a_frag(G + (size_t)m0 * 256 + kk, 256, lane);
    #pragma unroll
    for (int tt = 0; tt < 4; tt++) {
      frag_u bf = load_b_frag(WoT + (size_t)(nbase + tt * 16) * 256 + kk, 256, lane);
      acc[tt] = __builtin_amdgcn_wmma_f32_16x16x32_bf16(false, a.b, false, bf.b,
                                                        (short)0, acc[tt], false, false);
    }
  }
  int n = lane & 15, hf = lane >> 4;
  #pragma unroll
  for (int tt = 0; tt < 4; tt++) {
    int col = nbase + tt * 16 + n;
    float bov = bo[col];
    #pragma unroll
    for (int i = 0; i < 8; i++) {
      int t = m0 + i + 8 * hf;
      out[(size_t)t * 256 + col] = acc[tt][i] + bov;
    }
  }
}

// ---------------------------------------------------------------------------
extern "C" void kernel_launch(void* const* d_in, const int* in_sizes, int n_in,
                              void* d_out, int out_size, void* d_ws, size_t ws_size,
                              hipStream_t stream) {
  const float* x          = (const float*)d_in[0];
  const float* edges      = (const float*)d_in[1];
  const unsigned char* mask      = (const unsigned char*)d_in[2];
  const unsigned char* edge_mask = (const unsigned char*)d_in[3];
  const float* ln_g   = (const float*)d_in[4];
  const float* ln_b   = (const float*)d_in[5];
  const float* lne_g  = (const float*)d_in[6];
  const float* lne_b  = (const float*)d_in[7];
  const float* W_edge = (const float*)d_in[8];
  const float* Wq     = (const float*)d_in[9];
  const float* Wkv    = (const float*)d_in[10];
  const float* Wg     = (const float*)d_in[11];
  const float* bg     = (const float*)d_in[12];
  const float* Wo     = (const float*)d_in[13];
  const float* bo     = (const float*)d_in[14];
  float* out = (float*)d_out;

  char* ws = (char*)d_ws;
  size_t off = 0;
  auto alloc = [&](size_t bytes) -> void* {
    off = (off + 255) & ~(size_t)255;
    void* p = ws + off;
    off += bytes;
    return p;
  };
  unsigned short* xn   = (unsigned short*)alloc((size_t)TOK * 256 * 2);
  unsigned short* WT   = (unsigned short*)alloc((size_t)1024 * 256 * 2);
  unsigned short* WoT  = (unsigned short*)alloc((size_t)256 * 256 * 2);
  unsigned short* Qb   = (unsigned short*)alloc((size_t)TOK * 256 * 2);
  unsigned short* Kb   = (unsigned short*)alloc((size_t)TOK * 256 * 2);
  unsigned short* Vt   = (unsigned short*)alloc((size_t)TOK * 256 * 2);
  float*          gates= (float*)alloc((size_t)TOK * 256 * 4);
  float*          bias = (float*)alloc((size_t)H * WW * WW * 4);
  unsigned short* G    = (unsigned short*)alloc((size_t)TOK * 256 * 2);

  wconv_kernel   <<<1280,        256, 0, stream>>>(Wq, Wkv, Wg, Wo, WT, WoT);
  ln_nodes_kernel<<<TOK / 8,     256, 0, stream>>>(x, ln_g, ln_b, xn);
  edge_bias_kernel<<<(WW*WW)/8,  256, 0, stream>>>(edges, lne_g, lne_b, W_edge, edge_mask, bias);
  qkvg_kernel    <<<(2048*16)/8, 256, 0, stream>>>(xn, WT, bg, Qb, Kb, Vt, gates);
  attn_kernel    <<<RDIM * H,    512, 0, stream>>>(Qb, Kb, Vt, bias, mask, gates, G);
  out_gemm_kernel<<<(2048*4)/8,  256, 0, stream>>>(G, WoT, bo, out);
}
